// InfiniAttention_43207370998263
// MI455X (gfx1250) — matrix operand order repaired and checked by reading, not compiled, not run on hardware
//
#include <hip/hip_runtime.h>

// ---------------------------------------------------------------------------
// InfiniAttention for MI455X (gfx1250, wave32, WMMA).
// B=4, N=4096, D=1024, H=16, Dh=64, INNER=1024, SEG=32, W=128.
// ---------------------------------------------------------------------------

typedef __attribute__((ext_vector_type(16))) __bf16 v16bf;
typedef __attribute__((ext_vector_type(8)))  float  v8f;
typedef unsigned short u16;
typedef unsigned int   u32;

__device__ __forceinline__ u16 f2bf(float f) {
  u32 u = __float_as_uint(f);
  u32 r = ((u >> 16) & 1u) + 0x7fffu;   // round-to-nearest-even
  return (u16)((u + r) >> 16);
}
__device__ __forceinline__ float bf2f(u16 h) {
  return __uint_as_float(((u32)h) << 16);
}
__device__ __forceinline__ float elu1f(float x) {
  return x > 0.f ? x + 1.f : __expf(x);
}

// ---- WMMA fragment loaders (CDNA5 16-bit layouts, wave32) -----------------
// A (16x32, MxK): lane l holds M=l%16; VGPR p, pos q -> K = (p<4?0:16)+(p%4)*2+q + (l>=16?8:0).
// B (32x16, KxN) loaded from a TRANSPOSED LDS image BT[n][k] (k contiguous):
//   lane l holds N=l%16; slot j -> K = (j<8?0:16)+(j%8) + 8*(l/16).
union FragU { v16bf v; u32 u[8]; };

__device__ __forceinline__ v16bf load_frag_a(const u16* s, int stride, int m0, int k0, int lane) {
  FragU f;
  const u16* row = s + (m0 + (lane & 15)) * stride;
  const int kh = (lane >> 4) * 8;
#pragma unroll
  for (int p = 0; p < 8; ++p) {
    int k = k0 + kh + ((p & 3) * 2) + ((p >> 2) * 16);
    f.u[p] = *(const u32*)(row + k);
  }
  return f.v;
}
__device__ __forceinline__ v16bf load_frag_b(const u16* sT, int stride, int n0, int k0, int lane) {
  FragU f;
  const u16* row = sT + (n0 + (lane & 15)) * stride + k0 + (lane >> 4) * 8;
#pragma unroll
  for (int p = 0; p < 8; ++p) f.u[p] = *(const u32*)(row + ((p & 3) * 2) + ((p >> 2) * 16));
  return f.v;
}

// ---------------------------------------------------------------------------
// 1) LayerNorm (biased var, eps=1e-5) + bf16 cast.  One block per row, D=1024.
// ---------------------------------------------------------------------------
__global__ __launch_bounds__(256) void ln_bf16_kernel(
    const float* __restrict__ x, const float* __restrict__ g,
    const float* __restrict__ b, u16* __restrict__ xn) {
  const int row = blockIdx.x;
  const float* xr = x + (size_t)row * 1024;
  float v[4], s = 0.f, s2 = 0.f;
#pragma unroll
  for (int i = 0; i < 4; ++i) {
    v[i] = xr[threadIdx.x + i * 256];
    s += v[i]; s2 += v[i] * v[i];
  }
#pragma unroll
  for (int off = 16; off > 0; off >>= 1) {
    s  += __shfl_down(s,  off, 32);
    s2 += __shfl_down(s2, off, 32);
  }
  __shared__ float ws1[8], ws2[8];
  __shared__ float mu_s, rstd_s;
  if ((threadIdx.x & 31) == 0) { ws1[threadIdx.x >> 5] = s; ws2[threadIdx.x >> 5] = s2; }
  __syncthreads();
  if (threadIdx.x == 0) {
    float a = 0.f, a2 = 0.f;
#pragma unroll
    for (int i = 0; i < 8; ++i) { a += ws1[i]; a2 += ws2[i]; }
    float mu = a * (1.f / 1024.f);
    float var = a2 * (1.f / 1024.f) - mu * mu;
    mu_s = mu; rstd_s = rsqrtf(var + 1e-5f);
  }
  __syncthreads();
  const float mu = mu_s, rstd = rstd_s;
#pragma unroll
  for (int i = 0; i < 4; ++i) {
    int c = threadIdx.x + i * 256;
    xn[(size_t)row * 1024 + c] = f2bf((v[i] - mu) * rstd * g[c] + b[c]);
  }
}

// ---------------------------------------------------------------------------
// 2) fp32 -> bf16 elementwise
// ---------------------------------------------------------------------------
__global__ __launch_bounds__(256) void cvt_bf16_kernel(
    const float* __restrict__ in, u16* __restrict__ out, int n) {
  int i = blockIdx.x * 256 + threadIdx.x;
  if (i < n) out[i] = f2bf(in[i]);
}

// ---------------------------------------------------------------------------
// 3) bf16 WMMA GEMM: C[M,N] = A[M,K] @ B[K,N].  A,B bf16 row-major, C fp32.
//    Block 256 thr (8 waves), tile 128x128, K-step 32 (one WMMA K per step).
//    Register-staged software pipeline: next K-step's global loads issue right
//    after the barrier, hiding HBM/L2 latency behind the 8-WMMA burst.
// ---------------------------------------------------------------------------
#define LSTR 40  // LDS row stride in elements (80B: 16B-aligned, conflict-free)
__global__ __launch_bounds__(256) void gemm_bf16_kernel(
    const u16* __restrict__ A, const u16* __restrict__ Bm,
    float* __restrict__ C, int M, int N, int K) {
  __shared__ __align__(16) u16 As[128 * LSTR];
  __shared__ __align__(16) u16 Bs[128 * LSTR];
  const int t = threadIdx.x, lane = t & 31, w = t >> 5;
  const int m0 = blockIdx.x * 128, n0 = blockIdx.y * 128;
  const int wm = (w & 1) * 64, wn = (w >> 1) * 32;
  v8f acc[4][2];
#pragma unroll
  for (int i = 0; i < 4; ++i)
#pragma unroll
    for (int j = 0; j < 2; ++j) acc[i][j] = (v8f){};

  const int ar = t >> 1, ah = (t & 1) * 16;     // A: 2 thr/row, 16 elems each
  const int bk = t >> 3, bn = (t & 7) * 16;     // B: 8 thr/row, 16 elems each
  const int nk = K >> 5;

  uint4 ra0, ra1;
  union BU { uint4 q[2]; u16 s[16]; } rb;
  auto gload = [&](int kt2) {
    const u16* gA = A + (size_t)(m0 + ar) * K + kt2 * 32 + ah;
    __builtin_prefetch(gA + 32, 0, 1);                  // k-step+1 hint
    ra0 = *(const uint4*)gA;
    ra1 = *(const uint4*)(gA + 8);
    const u16* gB = Bm + (size_t)(kt2 * 32 + bk) * N + n0 + bn;
    __builtin_prefetch(gB + (size_t)32 * N, 0, 1);
    rb.q[0] = *(const uint4*)gB;
    rb.q[1] = *(const uint4*)(gB + 8);
  };

  gload(0);
  for (int kt = 0; kt < nk; ++kt) {
    // drain staged registers into LDS (A row-major, B transposed)
    *(uint4*)(As + ar * LSTR + ah)     = ra0;
    *(uint4*)(As + ar * LSTR + ah + 8) = ra1;
#pragma unroll
    for (int j = 0; j < 16; ++j) Bs[(bn + j) * LSTR + bk] = rb.s[j];
    __syncthreads();
    if (kt + 1 < nk) gload(kt + 1);   // overlap next global loads with WMMAs

    v16bf af[4], bf[2];
#pragma unroll
    for (int i = 0; i < 4; ++i) af[i] = load_frag_a(As, LSTR, wm + i * 16, 0, lane);
#pragma unroll
    for (int j = 0; j < 2; ++j) bf[j] = load_frag_b(Bs, LSTR, wn + j * 16, 0, lane);
#pragma unroll
    for (int i = 0; i < 4; ++i)
#pragma unroll
      for (int j = 0; j < 2; ++j)
        acc[i][j] = __builtin_amdgcn_wmma_f32_16x16x32_bf16(
            false, af[i], false, bf[j], (short)0, acc[i][j], false, false);
    __syncthreads();
  }
  // epilogue: C layout per lane: col = l%16, rows = 8*(l/16) + r
#pragma unroll
  for (int i = 0; i < 4; ++i)
#pragma unroll
    for (int j = 0; j < 2; ++j) {
      int col  = n0 + wn + j * 16 + (lane & 15);
      int rowb = m0 + wm + i * 16 + 8 * (lane >> 4);
#pragma unroll
      for (int r = 0; r < 8; ++r)
        C[(size_t)(rowb + r) * N + col] = acc[i][j][r];
    }
}

// ---------------------------------------------------------------------------
// 4) RoPE + elu1 + layout prep.  Block = (b,n); 256 thr x 4 elems (2 pairs).
//    qkv fp32 (B*N, 3072), cols [0,1024)=q [1024,2048)=k [2048,3072)=v.
//    Writes next_k/next_v fp32 (b,h,n,d) to d_out; bf16 scan tensors to ws.
// ---------------------------------------------------------------------------
__global__ __launch_bounds__(256) void prep_kernel(
    const float* __restrict__ qkv,
    float* __restrict__ nk, float* __restrict__ nv,
    u16* __restrict__ qrb, u16* __restrict__ krb,
    u16* __restrict__ sqb, u16* __restrict__ skb, u16* __restrict__ vbb) {
  const int blk = blockIdx.x;
  const int b = blk >> 12, n = blk & 4095;
  const float* rowp = qkv + (size_t)blk * 3072;
  const int i0 = threadIdx.x * 4;
  const float C0 = 9.210340371976184f / 64.f;  // ln(10000)/64
#pragma unroll
  for (int pp = 0; pp < 2; ++pp) {
    int i = i0 + pp * 2;                 // even index within (h,d) flat 1024
    int h = i >> 6, dd = i & 63;         // dd even
    float fr  = __expf(-(float)dd * C0);
    float ang = (float)n * fr;
    float cs = cosf(ang), sn = sinf(ang);
    float q0 = rowp[i],        q1 = rowp[i + 1];
    float k0 = rowp[1024 + i], k1 = rowp[1024 + i + 1];
    float v0 = rowp[2048 + i], v1 = rowp[2048 + i + 1];
    size_t ob = ((size_t)(b * 16 + h) * 4096 + n) * 64 + dd;
    nk[ob] = k0; nk[ob + 1] = k1;        // pre-rotary cache
    nv[ob] = v0; nv[ob + 1] = v1;
    qrb[ob]     = f2bf(q0 * cs - q1 * sn);
    qrb[ob + 1] = f2bf(q1 * cs + q0 * sn);
    krb[ob]     = f2bf(k0 * cs - k1 * sn);
    krb[ob + 1] = f2bf(k1 * cs + k0 * sn);
    sqb[ob]     = f2bf(elu1f(q0)); sqb[ob + 1] = f2bf(elu1f(q1));
    skb[ob]     = f2bf(elu1f(k0)); skb[ob + 1] = f2bf(elu1f(k1));
    vbb[ob]     = f2bf(v0);        vbb[ob + 1] = f2bf(v1);
  }
}

// ---------------------------------------------------------------------------
// 5) Segment scan.  One workgroup per (b,h); 4 waves; 128 segments of 32.
//    memT[v][k] = mem[k][v] kept fp32 (master) + bf16 (WMMA operand) in LDS.
//    64 WMMAs per segment; next segment's tiles are register-staged (b128
//    loads) so global latency hides behind the current segment's compute.
// ---------------------------------------------------------------------------
__global__ __launch_bounds__(128) void scan_kernel(
    const u16* __restrict__ qr, const u16* __restrict__ kr,
    const u16* __restrict__ sq, const u16* __restrict__ sk,
    const u16* __restrict__ vb, const float* __restrict__ mem_beta,
    u16* __restrict__ obuf) {
  const int bh = blockIdx.x;
  const int b = bh >> 4, h = bh & 15;
  const int t = threadIdx.x, lane = t & 31, w = t >> 5;
  const float beta = 1.f / (1.f + __expf(-mem_beta[h]));

  __shared__ __align__(16) float memf[64 * 64];   // memT fp32
  __shared__ __align__(16) u16   memb[64 * 64];   // memT bf16
  __shared__ float zsh[64];
  __shared__ __align__(16) u16 qt [32 * 64], kt [32 * 64], sqt[32 * 64],
                               skt[32 * 64], vt [32 * 64];
  __shared__ __align__(16) u16 sktT[64 * 32], vtT[64 * 32], dvT[64 * 32];
  __shared__ float sc[32 * 32];
  __shared__ __align__(16) u16 scb[32 * 32];
  __shared__ float sd[32 * 64];
  __shared__ float pn[32 * 64];
  __shared__ float denq[32], denk[32];

  for (int i = t; i < 64 * 64; i += 128) { memf[i] = 0.f; memb[i] = 0; }
  if (t < 64) zsh[t] = 1e-6f;
  __syncthreads();

  const size_t base = (size_t)bh * 4096 * 64;

  // register-staged tile loads: thread t owns 16 contiguous elems per tensor
  uint4 rg[5][2];
  auto gload_seg = [&](int s2) {
    const size_t off2 = base + (size_t)s2 * 2048 + (size_t)t * 16;
    rg[0][0] = *(const uint4*)(qr + off2); rg[0][1] = *(const uint4*)(qr + off2 + 8);
    rg[1][0] = *(const uint4*)(kr + off2); rg[1][1] = *(const uint4*)(kr + off2 + 8);
    rg[2][0] = *(const uint4*)(sq + off2); rg[2][1] = *(const uint4*)(sq + off2 + 8);
    rg[3][0] = *(const uint4*)(sk + off2); rg[3][1] = *(const uint4*)(sk + off2 + 8);
    rg[4][0] = *(const uint4*)(vb + off2); rg[4][1] = *(const uint4*)(vb + off2 + 8);
  };
  const int lm = t >> 2, ld0 = (t & 3) * 16;      // row, col base within 32x64
  auto lstore_seg = [&]() {
    *(uint4*)(qt  + lm * 64 + ld0)     = rg[0][0];
    *(uint4*)(qt  + lm * 64 + ld0 + 8) = rg[0][1];
    *(uint4*)(kt  + lm * 64 + ld0)     = rg[1][0];
    *(uint4*)(kt  + lm * 64 + ld0 + 8) = rg[1][1];
    *(uint4*)(sqt + lm * 64 + ld0)     = rg[2][0];
    *(uint4*)(sqt + lm * 64 + ld0 + 8) = rg[2][1];
    *(uint4*)(skt + lm * 64 + ld0)     = rg[3][0];
    *(uint4*)(skt + lm * 64 + ld0 + 8) = rg[3][1];
    *(uint4*)(vt  + lm * 64 + ld0)     = rg[4][0];
    *(uint4*)(vt  + lm * 64 + ld0 + 8) = rg[4][1];
    union { uint4 q[2]; u16 s[16]; } us, uv;
    us.q[0] = rg[3][0]; us.q[1] = rg[3][1];
    uv.q[0] = rg[4][0]; uv.q[1] = rg[4][1];
#pragma unroll
    for (int j = 0; j < 16; ++j) {
      sktT[(ld0 + j) * 32 + lm] = us.s[j];
      vtT [(ld0 + j) * 32 + lm] = uv.s[j];
    }
  };

  gload_seg(0);
  for (int seg = 0; seg < 128; ++seg) {
    lstore_seg();
    __syncthreads();
    if (seg + 1 < 128) gload_seg(seg + 1);   // hide next seg's HBM latency

    // ---- scores = (qr @ kr^T) * SCALE : 2x2 tiles, K=64 (2 steps) ----
    {
      int mi = w & 1, ni = w >> 1;
      v8f c = {};
#pragma unroll
      for (int ks = 0; ks < 2; ++ks) {
        v16bf a  = load_frag_a(qt, 64, mi * 16, ks * 32, lane);
        v16bf bf = load_frag_b(kt, 64, ni * 16, ks * 32, lane);  // kt rows are B^T rows
        c = __builtin_amdgcn_wmma_f32_16x16x32_bf16(false, a, false, bf, (short)0, c, false, false);
      }
      int col = ni * 16 + (lane & 15), rowb = mi * 16 + 8 * (lane >> 4);
#pragma unroll
      for (int r = 0; r < 8; ++r) sc[(rowb + r) * 32 + col] = c[r] * 0.125f;
    }
    __syncthreads();

    // ---- causal softmax (thr 0-31) || den_q (32-63) || den_k (64-95) ----
    if (t < 32) {
      int row = t; float mx = -1e30f;
      for (int j = 0; j <= row; ++j) mx = fmaxf(mx, sc[row * 32 + j]);
      float sum = 0.f;
      for (int j = 0; j < 32; ++j) {
        float e = (j <= row) ? __expf(sc[row * 32 + j] - mx) : 0.f;
        sc[row * 32 + j] = e; sum += e;
      }
      float inv = 1.f / sum;
      for (int j = 0; j < 32; ++j) scb[row * 32 + j] = f2bf(sc[row * 32 + j] * inv);
    } else if (t < 64) {
      int row = t - 32; float a = 0.f;
      for (int kk = 0; kk < 64; ++kk) a += bf2f(sqt[row * 64 + kk]) * zsh[kk];
      denq[row] = a;
    } else if (t < 96) {
      int row = t - 64; float a = 0.f;
      for (int kk = 0; kk < 64; ++kk) a += bf2f(skt[row * 64 + kk]) * zsh[kk];
      denk[row] = a;
    }
    __syncthreads();

    // ---- sdpa = probs @ v : 2x4 tiles, K=32 ----
#pragma unroll
    for (int i2 = 0; i2 < 2; ++i2) {
      int id = w * 2 + i2, mi = id & 1, nj = id >> 1;
      v16bf a  = load_frag_a(scb, 32, mi * 16, 0, lane);
      v16bf bf = load_frag_b(vtT, 32, nj * 16, 0, lane);
      v8f c = {};
      c = __builtin_amdgcn_wmma_f32_16x16x32_bf16(false, a, false, bf, (short)0, c, false, false);
      int col = nj * 16 + (lane & 15), rowb = mi * 16 + 8 * (lane >> 4);
#pragma unroll
      for (int r = 0; r < 8; ++r) sd[(rowb + r) * 64 + col] = c[r];
    }
    // ---- num = sq @ mem : 2x4 tiles, K=64 (memb rows are B^T rows) ----
#pragma unroll
    for (int i2 = 0; i2 < 2; ++i2) {
      int id = w * 2 + i2, mi = id & 1, nj = id >> 1;
      v8f c = {};
#pragma unroll
      for (int ks = 0; ks < 2; ++ks) {
        v16bf a  = load_frag_a(sqt, 64, mi * 16, ks * 32, lane);
        v16bf bf = load_frag_b(memb, 64, nj * 16, ks * 32, lane);
        c = __builtin_amdgcn_wmma_f32_16x16x32_bf16(false, a, false, bf, (short)0, c, false, false);
      }
      int col = nj * 16 + (lane & 15), rowb = mi * 16 + 8 * (lane >> 4);
#pragma unroll
      for (int r = 0; r < 8; ++r) pn[(rowb + r) * 64 + col] = c[r];
    }
    __syncthreads();

    // ---- out = beta * num/(den+eps) + (1-beta) * sdpa -> obuf (bf16) ----
    for (int i = t; i < 2048; i += 128) {
      int m = i >> 6, d = i & 63;
      float mo = pn[i] / (denq[m] + 1e-6f);
      float o = beta * mo + (1.f - beta) * sd[i];
      int n = seg * 32 + m;
      obuf[((size_t)(b * 4096 + n)) * 1024 + h * 64 + d] = f2bf(o);
    }
    __syncthreads();

    // ---- prev_num = sk @ mem : 2x4 tiles, K=64 ----
#pragma unroll
    for (int i2 = 0; i2 < 2; ++i2) {
      int id = w * 2 + i2, mi = id & 1, nj = id >> 1;
      v8f c = {};
#pragma unroll
      for (int ks = 0; ks < 2; ++ks) {
        v16bf a  = load_frag_a(skt, 64, mi * 16, ks * 32, lane);
        v16bf bf = load_frag_b(memb, 64, nj * 16, ks * 32, lane);
        c = __builtin_amdgcn_wmma_f32_16x16x32_bf16(false, a, false, bf, (short)0, c, false, false);
      }
      int col = nj * 16 + (lane & 15), rowb = mi * 16 + 8 * (lane >> 4);
#pragma unroll
      for (int r = 0; r < 8; ++r) pn[(rowb + r) * 64 + col] = c[r];
    }
    __syncthreads();

    // ---- dv = v - prev_num/(den_k+eps), build dvT (bf16) ----
    for (int i = t; i < 2048; i += 128) {
      int m = i >> 6, d = i & 63;
      float dv = bf2f(vt[i]) - pn[i] / (denk[m] + 1e-6f);
      dvT[d * 32 + m] = f2bf(dv);
    }
    __syncthreads();

    // ---- memT[v][k] += dv^T @ sk : A=dvT(64x32), B^T=sktT, 4x4 tiles K=32 ----
#pragma unroll
    for (int i2 = 0; i2 < 4; ++i2) {
      int id = w * 4 + i2, mi = id & 3, nj = id >> 2;
      v16bf a  = load_frag_a(dvT, 32, mi * 16, 0, lane);
      v16bf bf = load_frag_b(sktT, 32, nj * 16, 0, lane);
      v8f c = {};
      c = __builtin_amdgcn_wmma_f32_16x16x32_bf16(false, a, false, bf, (short)0, c, false, false);
      int col = nj * 16 + (lane & 15), rowb = mi * 16 + 8 * (lane >> 4);
#pragma unroll
      for (int r = 0; r < 8; ++r) memf[(rowb + r) * 64 + col] += c[r];
    }
    __syncthreads();

    // ---- refresh bf16 mem image; z += colsum(sk) ----
    for (int i = t; i < 4096; i += 128) memb[i] = f2bf(memf[i]);
    if (t < 64) {
      float a = 0.f;
      for (int m = 0; m < 32; ++m) a += bf2f(skt[m * 64 + t]);
      zsh[t] += a;
    }
    __syncthreads();
  }
}

// ---------------------------------------------------------------------------
// Launch
// ---------------------------------------------------------------------------
extern "C" void kernel_launch(void* const* d_in, const int* in_sizes, int n_in,
                              void* d_out, int out_size, void* d_ws, size_t ws_size,
                              hipStream_t stream) {
  (void)in_sizes; (void)n_in; (void)out_size; (void)ws_size;
  const float* x        = (const float*)d_in[0];
  const float* w_qkv    = (const float*)d_in[1];
  const float* w_out    = (const float*)d_in[2];
  const float* ln_g     = (const float*)d_in[3];
  const float* ln_b     = (const float*)d_in[4];
  const float* mem_beta = (const float*)d_in[5];

  float* out    = (float*)d_out;                 // (B,N,D)      16,777,216
  float* next_k = out + 16777216ull;             // (B,H,N,Dh)   16,777,216
  float* next_v = next_k + 16777216ull;          // (B,H,N,Dh)   16,777,216

  char* ws = (char*)d_ws;
  size_t o = 0;
  auto alloc = [&](size_t bytes) { char* p = ws + o; o += (bytes + 255) & ~(size_t)255; return p; };
  u16*   xn   = (u16*)  alloc(16384ull * 1024 * 2);   // layernormed x, bf16
  u16*   wqb  = (u16*)  alloc(1024ull * 3072 * 2);    // w_qkv bf16
  u16*   wob  = (u16*)  alloc(1024ull * 1024 * 2);    // w_out bf16
  float* qkv  = (float*)alloc(16384ull * 3072 * 4);   // qkv fp32
  u16*   qrb  = (u16*)  alloc(16777216ull * 2);       // rotary q bf16 (b,h,n,d)
  u16*   krb  = (u16*)  alloc(16777216ull * 2);       // rotary k bf16
  u16*   sqb  = (u16*)  alloc(16777216ull * 2);       // elu1(q) bf16
  u16*   skb  = (u16*)  alloc(16777216ull * 2);       // elu1(k) bf16
  u16*   vbb  = (u16*)  alloc(16777216ull * 2);       // v bf16
  u16*   obuf = (u16*)  alloc(16777216ull * 2);       // attention out bf16 (B,N,INNER)

  ln_bf16_kernel<<<16384, 256, 0, stream>>>(x, ln_g, ln_b, xn);
  cvt_bf16_kernel<<<(3145728 + 255) / 256, 256, 0, stream>>>(w_qkv, wqb, 3145728);
  cvt_bf16_kernel<<<(1048576 + 255) / 256, 256, 0, stream>>>(w_out, wob, 1048576);

  // qkv = xn @ w_qkv : 16384 x 3072 x 1024
  gemm_bf16_kernel<<<dim3(128, 24), 256, 0, stream>>>(xn, wqb, qkv, 16384, 3072, 1024);

  prep_kernel<<<16384, 256, 0, stream>>>(qkv, next_k, next_v, qrb, krb, sqb, skb, vbb);

  scan_kernel<<<64, 128, 0, stream>>>(qrb, krb, sqb, skb, vbb, mem_beta, obuf);

  // out = o @ w_out : 16384 x 1024 x 1024
  gemm_bf16_kernel<<<dim3(128, 8), 256, 0, stream>>>(obuf, wob, out, 16384, 1024, 1024);
}
